// TransformerEncoder_72739566125178
// MI455X (gfx1250) — matrix operand order repaired and checked by reading, not compiled, not run on hardware
//
#include <hip/hip_runtime.h>
#include <math.h>
#include <stdint.h>

// ---------------------------------------------------------------------------
// Problem constants (reference: B=64, S=197, D=768, H=4*D)
// ---------------------------------------------------------------------------
#define BATCH   64
#define S_LEN   197
#define D_EMB   768
#define H_DIM   3072
#define BS_TOK  (BATCH * S_LEN)      // 12608 tokens, divisible by 64
#define S_PAD   224                  // t padded to multiple of 32 for P@V
#define NT_S    13                   // ceil(197/16)
#define LN_EPS  1e-5f
// SCALE = 1 / S**(-0.5) = sqrt(S)  (faithful to source)
#define ATTN_SCALE 14.0356688476182f

// ---------------------------------------------------------------------------
// Vector / WMMA types
// ---------------------------------------------------------------------------
typedef __attribute__((ext_vector_type(16))) __bf16    v16bf;
typedef __attribute__((ext_vector_type(8)))  float     v8f;
typedef __attribute__((ext_vector_type(4)))  uint32_t  u32x4;
typedef __attribute__((ext_vector_type(4)))  float     f32x4;

union Frag16 { v16bf v; uint32_t u[8]; u32x4 q[2]; };

static __device__ inline v8f wmma_bf16(const Frag16& a, const Frag16& b, v8f c) {
    // (neg_a, A, neg_b, B, c_mod, C, reuse_a, reuse_b)
    return __builtin_amdgcn_wmma_f32_16x16x32_bf16(
        false, a.v, false, b.v, (short)0, c, false, false);
}

static __device__ inline unsigned short f2bf(float f) {
    union { float f; uint32_t u; } c; c.f = f;
    uint32_t u = c.u;
    u += 0x7FFFu + ((u >> 16) & 1u);   // round to nearest even
    return (unsigned short)(u >> 16);
}

static __device__ inline uint32_t pack2bf(float lo, float hi) {
    return (uint32_t)f2bf(lo) | ((uint32_t)f2bf(hi) << 16);
}

static __device__ inline float gelu_exact(float x) {
    return 0.5f * x * (1.0f + erff(x * 0.70710678118654752f));
}

// ---------------------------------------------------------------------------
// Weight transpose + f32->bf16 convert: WT[c*R + r] = bf16(W[r*C + c])
// ---------------------------------------------------------------------------
__global__ void convt_kernel(const float* __restrict__ W,
                             unsigned short* __restrict__ WT,
                             int R, int C) {
    int total = R * C;
    for (int idx = blockIdx.x * blockDim.x + threadIdx.x; idx < total;
         idx += gridDim.x * blockDim.x) {
        int r = idx / C, c = idx - r * C;
        WT[(size_t)c * R + r] = f2bf(W[(size_t)r * C + c]);
    }
}

// ---------------------------------------------------------------------------
// Batch-global LayerNorm over (S,D), elementwise affine (S,D); out = bf16
// One block per batch element; float4 loads, packed bf16x2 dword stores.
// ---------------------------------------------------------------------------
__global__ __launch_bounds__(256)
void layernorm_kernel(const float* __restrict__ x,
                      const float* __restrict__ w,
                      const float* __restrict__ bsh,
                      unsigned short* __restrict__ out) {
    __shared__ float rs[256], rs2[256];
    const int b = blockIdx.x, tid = threadIdx.x;
    const int N  = S_LEN * D_EMB;     // 151296 (divisible by 4)
    const int N4 = N / 4;
    const f32x4* xb4 = (const f32x4*)(x + (size_t)b * N);
    const f32x4* w4  = (const f32x4*)w;
    const f32x4* b4  = (const f32x4*)bsh;

    float s = 0.f, s2 = 0.f;
    for (int i = tid; i < N4; i += 256) {
        f32x4 v = xb4[i];
        s  += v[0] + v[1] + v[2] + v[3];
        s2 += v[0]*v[0] + v[1]*v[1] + v[2]*v[2] + v[3]*v[3];
    }
    rs[tid] = s; rs2[tid] = s2;
    __syncthreads();
    for (int st = 128; st > 0; st >>= 1) {
        if (tid < st) { rs[tid] += rs[tid + st]; rs2[tid] += rs2[tid + st]; }
        __syncthreads();
    }
    float mu   = rs[0] / (float)N;
    float var  = rs2[0] / (float)N - mu * mu;
    float rstd = rsqrtf(var + LN_EPS);
    uint32_t* ob32 = (uint32_t*)(out + (size_t)b * N);
    for (int i = tid; i < N4; i += 256) {
        f32x4 v = xb4[i], ww = w4[i], bb = b4[i];
        float y0 = (v[0] - mu) * rstd * ww[0] + bb[0];
        float y1 = (v[1] - mu) * rstd * ww[1] + bb[1];
        float y2 = (v[2] - mu) * rstd * ww[2] + bb[2];
        float y3 = (v[3] - mu) * rstd * ww[3] + bb[3];
        ob32[i * 2]     = pack2bf(y0, y1);
        ob32[i * 2 + 1] = pack2bf(y2, y3);
    }
}

// ---------------------------------------------------------------------------
// Tiled NT GEMM: C[m][n] = sum_k A[m][k] * Bt[n][k], A/Bt bf16 row-major.
// Block tile 64x128, K-step 32, 128 threads (4 waves).
// Each wave owns a 32x64 sub-panel: 2 A frags x 4 B frags = 8 WMMA / k-step.
// Staging uses 16B chunks: global_load_b128 -> ds_store_b128.
// Epilogue: 0 = +bias -> bf16 out
//           1 = +bias -> bf16 out AND scatter vT[(b*768+n)*224 + t]
//           2 = gelu(+bias) -> bf16 out
//           3 = gelu(+bias) + resid -> f32 out
// M multiple of 64; N multiple of 128; K multiple of 32 (no bounds checks).
// ---------------------------------------------------------------------------
__global__ __launch_bounds__(128)
void gemm_nt_bf16(const unsigned short* __restrict__ A,
                  const unsigned short* __restrict__ Bt,
                  int M, int N, int K,
                  const float* __restrict__ bias,
                  int epilogue,
                  unsigned short* __restrict__ outB,
                  unsigned short* __restrict__ vT,
                  const float* __restrict__ resid,
                  float* __restrict__ outF) {
    __shared__ uint32_t sA[64 * 16];    // 64 rows  x 32 bf16 (16 dwords/row)
    __shared__ uint32_t sB[128 * 16];   // 128 rows x 32 bf16

    const int tid  = threadIdx.x;
    const int lane = tid & 31;
    const int wave = tid >> 5;
    const int half = lane >> 4;
    const int l16  = lane & 15;
    const int mw   = wave >> 1;         // 0..1 : 32-row half
    const int nw   = wave & 1;          // 0..1 : 64-col half
    const int m0   = blockIdx.x * 64;
    const int n0   = blockIdx.y * 128;

    v8f acc[2][4];
    #pragma unroll
    for (int mi = 0; mi < 2; ++mi)
        #pragma unroll
        for (int nb = 0; nb < 4; ++nb) acc[mi][nb] = (v8f){0,0,0,0,0,0,0,0};

    const uint32_t* A32 = (const uint32_t*)A;
    const uint32_t* B32 = (const uint32_t*)Bt;
    u32x4* sA4 = (u32x4*)sA;
    u32x4* sB4 = (u32x4*)sB;
    const u32x4* sA4c = (const u32x4*)sA;
    const u32x4* sB4c = (const u32x4*)sB;
    const int Kd = K >> 1;  // dwords per row

    for (int k0 = 0; k0 < K; k0 += 32) {
        __syncthreads();
        const int kd = k0 >> 1;
        // A panel: 256 x 16B chunks, 2 per thread.
        #pragma unroll
        for (int j = 0; j < 2; ++j) {
            int c = tid + j * 128;                 // chunk id
            int row = c >> 2, col = (c & 3) * 4;   // 4 chunks per 16-dword row
            sA4[c] = *(const u32x4*)&A32[(size_t)(m0 + row) * Kd + kd + col];
        }
        // B panel: 512 x 16B chunks, 4 per thread.
        #pragma unroll
        for (int j = 0; j < 4; ++j) {
            int c = tid + j * 128;
            int row = c >> 2, col = (c & 3) * 4;
            sB4[c] = *(const u32x4*)&B32[(size_t)(n0 + row) * Kd + kd + col];
        }
        if (k0 + 32 < K) {  // prefetch next k-panel (global_prefetch_b8)
            __builtin_prefetch(&A32[(size_t)(m0 + (tid >> 1)) * Kd + kd + 16], 0, 1);
            __builtin_prefetch(&B32[(size_t)(n0 + (tid >> 1)) * Kd + kd + 16], 0, 1);
            __builtin_prefetch(&B32[(size_t)(n0 + 64 + (tid >> 1)) * Kd + kd + 16], 0, 1);
        }
        __syncthreads();

        Frag16 af[2];
        #pragma unroll
        for (int mi = 0; mi < 2; ++mi) {
            int rbase = (mw * 32 + mi * 16 + l16) * 4;  // row * 16 dwords / 4
            af[mi].q[0] = sA4c[rbase + 2 * half];
            af[mi].q[1] = sA4c[rbase + 2 * half + 1];
        }
        Frag16 bf[4];
        #pragma unroll
        for (int nb = 0; nb < 4; ++nb) {
            int rbase = (nw * 64 + nb * 16 + l16) * 4 + half * 2;
            bf[nb].q[0] = sB4c[rbase];
            bf[nb].q[1] = sB4c[rbase + 1];
        }

        #pragma unroll
        for (int mi = 0; mi < 2; ++mi)
            #pragma unroll
            for (int nb = 0; nb < 4; ++nb)
                acc[mi][nb] = wmma_bf16(af[mi], bf[nb], acc[mi][nb]);
    }

    #pragma unroll
    for (int mi = 0; mi < 2; ++mi) {
        #pragma unroll
        for (int nb = 0; nb < 4; ++nb) {
            int n = n0 + nw * 64 + nb * 16 + l16;
            float bv = bias[n];
            #pragma unroll
            for (int r = 0; r < 8; ++r) {
                int m = m0 + mw * 32 + mi * 16 + r + half * 8;
                float val = acc[mi][nb][r] + bv;
                if (epilogue <= 1) {
                    unsigned short hb = f2bf(val);
                    outB[(size_t)m * N + n] = hb;
                    if (epilogue == 1) {
                        int bb = m / S_LEN, t = m - bb * S_LEN;
                        vT[((size_t)bb * D_EMB + n) * S_PAD + t] = hb;
                    }
                } else {
                    float g = gelu_exact(val);
                    if (epilogue == 2) {
                        outB[(size_t)m * N + n] = f2bf(g);
                    } else {
                        size_t o = (size_t)m * N + n;
                        outF[o] = g + resid[o];
                    }
                }
            }
        }
    }
}

// ---------------------------------------------------------------------------
// Fused attention per (batch, 16-row tile):
//   scores = (Qtile @ K^T) * sqrt(S); softmax; out = P @ V; x1 = out + x
// Q/K bf16 row-major (BS x D); vT bf16 (B x D x S_PAD, zero-padded).
// One wave (32 threads) per block.
// ---------------------------------------------------------------------------
__global__ __launch_bounds__(32)
void attention_kernel(const unsigned short* __restrict__ Q,
                      const unsigned short* __restrict__ Kb,
                      const unsigned short* __restrict__ vT,
                      const float* __restrict__ x,
                      float* __restrict__ x1) {
    __shared__ uint32_t sQ[16 * (D_EMB / 2)];   // 16x768 bf16 = 24 KB
    __shared__ float    sS[16 * S_PAD];         // scores f32  = 14 KB
    __shared__ uint32_t sP[16 * (S_PAD / 2)];   // probs bf16  =  7 KB

    const int lane = threadIdx.x;
    const int half = lane >> 4;
    const int l16  = lane & 15;
    const int mt   = blockIdx.x;   // 0..12
    const int b    = blockIdx.y;   // 0..63
    const int s0   = mt * 16;

    // Stage the 16 x 768 Q tile in LDS (16B chunks; zero rows beyond S_LEN).
    const uint32_t* Q32 = (const uint32_t*)Q;
    u32x4* sQ4 = (u32x4*)sQ;
    const int QCH = D_EMB / 8;     // 96 16B-chunks per row
    for (int c = lane; c < 16 * QCH; c += 32) {
        int row = c / QCH, col = (c - row * QCH) * 4;
        int s = s0 + row;
        sQ4[c] = (s < S_LEN)
                 ? *(const u32x4*)&Q32[(size_t)(b * S_LEN + s) * (D_EMB / 2) + col]
                 : (u32x4){0, 0, 0, 0};
    }
    __syncthreads();

    // scores = Q @ K^T (B operand is K itself: N-major over t).
    const uint32_t* K32 = (const uint32_t*)Kb;
    const u32x4* sQ4c = (const u32x4*)sQ;
    for (int nt = 0; nt < NT_S; ++nt) {
        int t = nt * 16 + l16;
        bool tv = (t < S_LEN);
        size_t krow = (size_t)(b * S_LEN + (tv ? t : 0)) * (D_EMB / 2);
        v8f acc = (v8f){0,0,0,0,0,0,0,0};
        for (int ks = 0; ks < D_EMB / 32; ++ks) {
            Frag16 af, bf;
            int abase = (l16 * (D_EMB / 2) + ks * 16) >> 2;
            af.q[0] = sQ4c[abase + 2 * half];
            af.q[1] = sQ4c[abase + 2 * half + 1];
            if (tv) {
                bf.q[0] = *(const u32x4*)&K32[krow + ks * 16 + half * 8];
                bf.q[1] = *(const u32x4*)&K32[krow + ks * 16 + half * 8 + 4];
            } else {
                bf.q[0] = (u32x4){0,0,0,0};
                bf.q[1] = (u32x4){0,0,0,0};
            }
            acc = wmma_bf16(af, bf, acc);
        }
        #pragma unroll
        for (int r = 0; r < 8; ++r)
            sS[(r + half * 8) * S_PAD + nt * 16 + l16] = acc[r] * ATTN_SCALE;
    }
    __syncthreads();

    // Row softmax (lanes 0..15 each own one row), write bf16 P, zero-pad t.
    if (lane < 16) {
        float mx = -1e30f;
        for (int t = 0; t < S_LEN; ++t) mx = fmaxf(mx, sS[lane * S_PAD + t]);
        float sum = 0.f;
        for (int t = 0; t < S_LEN; ++t) {
            float e = __expf(sS[lane * S_PAD + t] - mx);
            sS[lane * S_PAD + t] = e; sum += e;
        }
        float inv = 1.0f / sum;
        unsigned short* sP16 = (unsigned short*)sP;
        for (int t = 0; t < S_LEN; ++t) sP16[lane * S_PAD + t] = f2bf(sS[lane * S_PAD + t] * inv);
        for (int t = S_LEN; t < S_PAD; ++t) sP16[lane * S_PAD + t] = 0;
    }
    __syncthreads();

    // out = P @ V via vT (N-major, zero-padded K=S_PAD); add residual x.
    const uint32_t* V32 = (const uint32_t*)vT;
    const u32x4* sP4c = (const u32x4*)sP;
    for (int nt2 = 0; nt2 < D_EMB / 16; ++nt2) {
        int n = nt2 * 16 + l16;
        size_t vrow = ((size_t)b * D_EMB + n) * (S_PAD / 2);
        v8f acc = (v8f){0,0,0,0,0,0,0,0};
        for (int ks = 0; ks < S_PAD / 32; ++ks) {
            Frag16 af, bf;
            int abase = (l16 * (S_PAD / 2) + ks * 16) >> 2;
            af.q[0] = sP4c[abase + 2 * half];
            af.q[1] = sP4c[abase + 2 * half + 1];
            bf.q[0] = *(const u32x4*)&V32[vrow + ks * 16 + half * 8];
            bf.q[1] = *(const u32x4*)&V32[vrow + ks * 16 + half * 8 + 4];
            acc = wmma_bf16(af, bf, acc);
        }
        #pragma unroll
        for (int r = 0; r < 8; ++r) {
            int s = s0 + r + half * 8;
            if (s < S_LEN) {
                size_t o = (size_t)(b * S_LEN + s) * D_EMB + n;
                x1[o] = acc[r] + x[o];
            }
        }
    }
}

// ---------------------------------------------------------------------------
// Host-side orchestration
// ---------------------------------------------------------------------------
static constexpr size_t alignup(size_t v) { return (v + 255) & ~(size_t)255; }

extern "C" void kernel_launch(void* const* d_in, const int* in_sizes, int n_in,
                              void* d_out, int out_size, void* d_ws, size_t ws_size,
                              hipStream_t stream) {
    (void)in_sizes; (void)n_in; (void)out_size; (void)ws_size;
    const float* x    = (const float*)d_in[0];
    const float* n1w  = (const float*)d_in[1];
    const float* n1b  = (const float*)d_in[2];
    const float* Wq   = (const float*)d_in[3];
    const float* bq   = (const float*)d_in[4];
    const float* Wk   = (const float*)d_in[5];
    const float* bk   = (const float*)d_in[6];
    const float* Wv   = (const float*)d_in[7];
    const float* bv   = (const float*)d_in[8];
    const float* n2w  = (const float*)d_in[9];
    const float* n2b  = (const float*)d_in[10];
    const float* W1   = (const float*)d_in[11];
    const float* b1   = (const float*)d_in[12];
    const float* W2   = (const float*)d_in[13];
    const float* b2   = (const float*)d_in[14];
    float* out = (float*)d_out;

    // Workspace layout (bf16 = 2B). hid overlaps the q/k/v/vT region.
    constexpr size_t SZ_H   = (size_t)BS_TOK * D_EMB * 2;        // 19,365,888
    constexpr size_t SZ_WT  = (size_t)D_EMB * D_EMB * 2;         //  1,179,648
    constexpr size_t SZ_W1T = (size_t)H_DIM * D_EMB * 2;         //  4,718,592
    constexpr size_t SZ_QKV = SZ_H;
    constexpr size_t SZ_VT  = (size_t)BATCH * D_EMB * S_PAD * 2; // 22,020,096
    constexpr size_t SZ_X1  = (size_t)BS_TOK * D_EMB * 4;        // 38,731,776

    char* ws = (char*)d_ws;
    size_t off = 0;
    unsigned short* h    = (unsigned short*)(ws + off); off = alignup(off + SZ_H);
    unsigned short* wqT  = (unsigned short*)(ws + off); off = alignup(off + SZ_WT);
    unsigned short* wkT  = (unsigned short*)(ws + off); off = alignup(off + SZ_WT);
    unsigned short* wvT  = (unsigned short*)(ws + off); off = alignup(off + SZ_WT);
    unsigned short* w1T  = (unsigned short*)(ws + off); off = alignup(off + SZ_W1T);
    unsigned short* w2T  = (unsigned short*)(ws + off); off = alignup(off + SZ_W1T);
    size_t qoff = off;
    unsigned short* qb   = (unsigned short*)(ws + off); off = alignup(off + SZ_QKV);
    unsigned short* kb   = (unsigned short*)(ws + off); off = alignup(off + SZ_QKV);
    unsigned short* vb   = (unsigned short*)(ws + off); off = alignup(off + SZ_QKV);
    unsigned short* vT   = (unsigned short*)(ws + off); off = alignup(off + SZ_VT);
    float*          x1   = (float*)(ws + off);          off = alignup(off + SZ_X1);
    unsigned short* hid  = (unsigned short*)(ws + qoff);  // reuse q/k/v/vT region
    unsigned short* h2   = h;                             // reuse h

    // vT pad columns must be zero (P@V reads K=S_PAD).
    hipMemsetAsync(vT, 0, SZ_VT, stream);

    // 1. Weight transpose + bf16 convert.
    convt_kernel<<<512, 256, 0, stream>>>(Wq, wqT, D_EMB, D_EMB);
    convt_kernel<<<512, 256, 0, stream>>>(Wk, wkT, D_EMB, D_EMB);
    convt_kernel<<<512, 256, 0, stream>>>(Wv, wvT, D_EMB, D_EMB);
    convt_kernel<<<1024, 256, 0, stream>>>(W1, w1T, D_EMB, H_DIM);
    convt_kernel<<<1024, 256, 0, stream>>>(W2, w2T, H_DIM, D_EMB);

    // 2. LN1 -> bf16 h.
    layernorm_kernel<<<BATCH, 256, 0, stream>>>(x, n1w, n1b, h);

    // 3. Q/K/V projections (M=12608, N=768, K=768).
    dim3 gproj(BS_TOK / 64, D_EMB / 128);
    gemm_nt_bf16<<<gproj, 128, 0, stream>>>(h, wqT, BS_TOK, D_EMB, D_EMB, bq, 0,
                                            qb, nullptr, nullptr, nullptr);
    gemm_nt_bf16<<<gproj, 128, 0, stream>>>(h, wkT, BS_TOK, D_EMB, D_EMB, bk, 0,
                                            kb, nullptr, nullptr, nullptr);
    gemm_nt_bf16<<<gproj, 128, 0, stream>>>(h, wvT, BS_TOK, D_EMB, D_EMB, bv, 1,
                                            vb, vT, nullptr, nullptr);

    // 4. Fused attention + residual -> x1.
    attention_kernel<<<dim3(NT_S, BATCH), 32, 0, stream>>>(qb, kb, vT, x, x1);

    // 5. LN2 -> bf16 h2.
    layernorm_kernel<<<BATCH, 256, 0, stream>>>(x1, n2w, n2b, h2);

    // 6. MLP1: gelu(h2 @ W1 + b1) -> bf16 hid (M=12608, N=3072, K=768).
    gemm_nt_bf16<<<dim3(BS_TOK / 64, H_DIM / 128), 128, 0, stream>>>(
        h2, w1T, BS_TOK, H_DIM, D_EMB, b1, 2, hid, nullptr, nullptr, nullptr);

    // 7. MLP2: gelu(hid @ W2 + b2) + x1 -> out (M=12608, N=768, K=3072).
    gemm_nt_bf16<<<dim3(BS_TOK / 64, D_EMB / 128), 128, 0, stream>>>(
        hid, w2T, BS_TOK, D_EMB, H_DIM, b2, 3, nullptr, nullptr, x1, out);
}
